// PPNet_84954453115675
// MI455X (gfx1250) — matrix operand order repaired and checked by reading, not compiled, not run on hardware
//
#include <hip/hip_runtime.h>
#include <hip/hip_bf16.h>
#include <math.h>
#include <stdint.h>

typedef __attribute__((ext_vector_type(16))) _Float16 v16h;
typedef __attribute__((ext_vector_type(8)))  _Float16 v8h;
typedef __attribute__((ext_vector_type(8)))  float    v8f;
typedef __attribute__((ext_vector_type(4)))  int      v4i;

#define NPOS 12544   // B*H*W = 64*14*14
#define HWSZ 196
#define CFEAT 2048

union Frag16 { v16h v; v8h h[2]; };

// ---- CDNA5 async global->LDS path (ASYNCcnt) ----
#if defined(__HIP_DEVICE_COMPILE__) && __has_builtin(__builtin_amdgcn_global_load_async_to_lds_b128)
#define USE_ASYNC 1
#else
#define USE_ASYNC 0
#endif

static __device__ __forceinline__ void async_cp_b128(const void* g, void* l)
{
#if USE_ASYNC
  __attribute__((address_space(1))) v4i* gp =
      (__attribute__((address_space(1))) v4i*)(uintptr_t)g;
  __attribute__((address_space(3))) v4i* lp =
      (__attribute__((address_space(3))) v4i*)(unsigned int)(uintptr_t)l;
  __builtin_amdgcn_global_load_async_to_lds_b128(gp, lp, 0, 0);
#endif
}

static __device__ __forceinline__ void wait_async0()
{
#if __has_builtin(__builtin_amdgcn_s_wait_asynccnt)
  __builtin_amdgcn_s_wait_asynccnt(0);
#else
  asm volatile("s_wait_asynccnt 0x0" ::: "memory");
#endif
}

// ----------------------------------------------------------------------------
// WMMA GEMM:  Out[M x Cout] = epilogue( A[M x Cin] * W^T + ... )
// MODE 0: ReLU(acc+bias)      -> f16 out
// MODE 1: sigmoid(acc+bias)   -> f16 out
// MODE 2: relu(x2[m]-2*acc+p2[n]) -> f32 out  (l2-distance epilogue)
// XSRC  : A is the raw input x [B,2048,14,14] f32, transposed+converted on load
// Block: 256 threads (8 waves). Tile M=128, N=64, K-step=32.
// ----------------------------------------------------------------------------
template<int MODE, bool XSRC>
__global__ __launch_bounds__(256)
void ppnet_gemm_wmma(const void* __restrict__ Aptr, const _Float16* __restrict__ W,
                     const float* __restrict__ bias, void* __restrict__ Out,
                     const float* __restrict__ x2, const float* __restrict__ p2,
                     int Cin, int Cout)
{
  __shared__ __align__(16) _Float16 As[128][40];
  __shared__ __align__(16) _Float16 Bs[64][40];

  const int m0   = blockIdx.x * 128;
  const int n0   = blockIdx.y * 64;
  const int t    = threadIdx.x;
  const int wid  = t >> 5;
  const int lane = t & 31;
  const int wm   = (wid & 3) * 32;   // wave row offset in block tile
  const int wn   = (wid >> 2) * 32;  // wave col offset in block tile
  const int half = lane >> 4;        // lane group 0/1
  const int lr   = lane & 15;

  // hoisted per-thread staging indices
  const int xm    = t & 127;           // XSRC: row within tile
  const int xkh   = (t >> 7) * 16;     // XSRC: 16 consecutive k per thread
  const int am    = t >> 1;            // f16 A stage: row
  const int akk   = (t & 1) * 16;      // f16 A stage: 16 consecutive k
  const int bn    = t >> 2;            // B stage: row (output channel)
  const int bkk   = (t & 3) * 8;       // B stage: 8 consecutive k

  // XSRC base pointer (constant per thread; k-loop adds immediate offsets)
  const float* xbase = nullptr;
  if (XSRC) {
    int row = m0 + xm;
    int b   = row / HWSZ;
    int hw  = row - b * HWSZ;
    xbase = (const float*)Aptr + (size_t)b * CFEAT * HWSZ + (size_t)xkh * HWSZ + hw;
  }

  v8f acc[2][2] = {};

  for (int k0 = 0; k0 < Cin; k0 += 32) {
    // ---- stage A tile (128 x 32 halves) ----
    if (XSRC) {
      // 16 pipelined loads into registers, convert, 2 vector LDS stores
      const float* xp = xbase + (size_t)k0 * HWSZ;
      float tmp[16];
      #pragma unroll
      for (int i = 0; i < 16; ++i) tmp[i] = xp[(size_t)i * HWSZ];
      Frag16 f;
      #pragma unroll
      for (int i = 0; i < 8; ++i) {
        f.h[0][i] = (_Float16)tmp[i];
        f.h[1][i] = (_Float16)tmp[8 + i];
      }
      *(v8h*)&As[xm][xkh]     = f.h[0];
      *(v8h*)&As[xm][xkh + 8] = f.h[1];
    } else {
      const _Float16* A = (const _Float16*)Aptr;
      const _Float16* src = A + (size_t)(m0 + am) * Cin + k0 + akk;
#if USE_ASYNC
      async_cp_b128(src,     &As[am][akk]);
      async_cp_b128(src + 8, &As[am][akk + 8]);
#else
      *(v8h*)&As[am][akk]     = *(const v8h*)(src);
      *(v8h*)&As[am][akk + 8] = *(const v8h*)(src + 8);
      if (k0 + 32 < Cin) __builtin_prefetch(src + 32, 0, 1);
#endif
    }
    // ---- stage B tile transposed: Bs[n][k] = W[n0+n][k0+k] ----
    {
      const _Float16* wsrc = W + (size_t)(n0 + bn) * Cin + k0 + bkk;
#if USE_ASYNC
      async_cp_b128(wsrc, &Bs[bn][bkk]);
#else
      *(v8h*)&Bs[bn][bkk] = *(const v8h*)(wsrc);
      if (k0 + 32 < Cin) __builtin_prefetch(wsrc + 32, 0, 1);
#endif
    }
#if USE_ASYNC
    wait_async0();
#endif
    __syncthreads();

    // ---- load fragments per documented wave32 layouts ----
    // A 16x32 f16: lanes 0-15 -> K 0..7 / 16..23 ; lanes 16-31 -> K 8..15 / 24..31
    Frag16 afr[2], bfr[2];
    #pragma unroll
    for (int mi = 0; mi < 2; ++mi) {
      const int m = wm + mi * 16 + lr;
      afr[mi].h[0] = *(const v8h*)&As[m][half * 8];
      afr[mi].h[1] = *(const v8h*)&As[m][16 + half * 8];
    }
    // B 32x16 f16: lanes 0-15 -> K 0..7 / 16..23 ; lanes 16-31 -> K 8..15 / 24..31
    #pragma unroll
    for (int ni = 0; ni < 2; ++ni) {
      const int n = wn + ni * 16 + lr;
      bfr[ni].h[0] = *(const v8h*)&Bs[n][half * 8];
      bfr[ni].h[1] = *(const v8h*)&Bs[n][16 + half * 8];
    }
    #pragma unroll
    for (int mi = 0; mi < 2; ++mi)
      #pragma unroll
      for (int ni = 0; ni < 2; ++ni)
        acc[mi][ni] = __builtin_amdgcn_wmma_f32_16x16x32_f16(
            false, afr[mi].v, false, bfr[ni].v, (short)0, acc[mi][ni], false, false);
    __syncthreads();
  }

  // ---- epilogue + store ----
  // C/D layout: lanes 0-15: M = vgpr(+0), N = lane ; lanes 16-31: M = vgpr+8
  #pragma unroll
  for (int mi = 0; mi < 2; ++mi) {
    #pragma unroll
    for (int ni = 0; ni < 2; ++ni) {
      #pragma unroll
      for (int r = 0; r < 8; ++r) {
        int row = m0 + wm + mi * 16 + half * 8 + r;
        int col = n0 + wn + ni * 16 + lr;
        float v = acc[mi][ni][r];
        if (MODE == 0) {
          v += bias[col];
          v = v > 0.f ? v : 0.f;
          ((_Float16*)Out)[(size_t)row * Cout + col] = (_Float16)v;
        } else if (MODE == 1) {
          v += bias[col];
          v = 1.f / (1.f + expf(-v));
          ((_Float16*)Out)[(size_t)row * Cout + col] = (_Float16)v;
        } else {
          float d = x2[row] - 2.f * v + p2[col];
          d = d > 0.f ? d : 0.f;
          ((float*)Out)[(size_t)row * Cout + col] = d;
        }
      }
    }
  }
}

// ----------------------------------------------------------------------------
__global__ void ppnet_cvt_f16(const float* __restrict__ src, _Float16* __restrict__ dst, int n)
{
  int i = blockIdx.x * 256 + threadIdx.x;
  if (i < n) dst[i] = (_Float16)src[i];
}

// protosT f16 [256][128] zero-padded rows >= 200
__global__ void ppnet_build_protos(const float* __restrict__ protos, _Float16* __restrict__ pt)
{
  int i = blockIdx.x * 256 + threadIdx.x;   // over 256*128
  int p = i >> 7, c = i & 127;
  float v = (p < 200) ? protos[p * 128 + c] : 0.f;
  pt[i] = (_Float16)v;
}

__global__ void ppnet_build_p2(const float* __restrict__ protos, float* __restrict__ p2)
{
  int p = threadIdx.x;                      // 256 threads
  float s = 0.f;
  if (p < 200)
    for (int c = 0; c < 128; ++c) { float v = protos[p * 128 + c]; s += v * v; }
  p2[p] = s;
}

// per-position sum of h^2  (h: [NPOS][128] f16)
__global__ void ppnet_x2(const _Float16* __restrict__ h, float* __restrict__ x2)
{
  int n = blockIdx.x * 256 + threadIdx.x;
  if (n >= NPOS) return;
  const _Float16* r = h + (size_t)n * 128;
  float s = 0.f;
  #pragma unroll 8
  for (int c = 0; c < 128; ++c) { float v = (float)r[c]; s += v * v; }
  x2[n] = s;
}

// min over 196 positions per (batch, prototype); also distance->similarity
__global__ void ppnet_minpool(const float* __restrict__ dist, float* __restrict__ sim,
                              float* __restrict__ out_mind)
{
  int b  = blockIdx.x;        // 64
  int kp = threadIdx.x;       // 256, use first 200
  if (kp >= 200) return;
  const float* dp = dist + (size_t)b * HWSZ * 256 + kp;
  float m = 3.402823466e38f;
  for (int hw = 0; hw < HWSZ; ++hw) { float v = dp[(size_t)hw * 256]; m = v < m ? v : m; }
  out_mind[b * 200 + kp] = m;
  sim[b * 200 + kp] = logf((m + 1.0f) / (m + 1e-4f));
}

// task heads + fc1 + batch-stat BN + ReLU + fc2 + sigmoid (single block)
__global__ void ppnet_heads(const float* __restrict__ sim,
                            const float* __restrict__ task_w, const float* __restrict__ task_b,
                            const float* __restrict__ fc1_w,  const float* __restrict__ fc1_b,
                            const float* __restrict__ bn_g,   const float* __restrict__ bn_b,
                            const float* __restrict__ fc2_w,  const float* __restrict__ fc2_b,
                            float* __restrict__ out_final, float* __restrict__ out_task)
{
  __shared__ float ys[64][20];
  __shared__ float mu[20], iv[20];
  int t = threadIdx.x;

  // task logits: [B,K,2] flat = b*20 + k*2 + c
  for (int o = t; o < 1280; o += 256) {
    int b = o / 20, rem = o % 20, k = rem >> 1, c = rem & 1;
    float s = task_b[k * 2 + c];
    for (int p = 0; p < 20; ++p)
      s += sim[b * 200 + k * 20 + p] * task_w[(k * 2 + c) * 20 + p];
    out_task[o] = s;
  }
  // fc1: y[64][20]
  for (int o = t; o < 1280; o += 256) {
    int b = o / 20, f = o % 20;
    float s = fc1_b[f];
    for (int i = 0; i < 200; ++i) s += sim[b * 200 + i] * fc1_w[f * 200 + i];
    ys[b][f] = s;
  }
  __syncthreads();
  if (t < 20) {
    float m = 0.f;
    for (int b = 0; b < 64; ++b) m += ys[b][t];
    m *= (1.f / 64.f);
    float v = 0.f;
    for (int b = 0; b < 64; ++b) { float d = ys[b][t] - m; v += d * d; }
    v *= (1.f / 64.f);
    mu[t] = m; iv[t] = rsqrtf(v + 1e-5f);
  }
  __syncthreads();
  for (int o = t; o < 1280; o += 256) {
    int b = o / 20, f = o % 20;
    float z = (ys[b][f] - mu[f]) * iv[f] * bn_g[f] + bn_b[f];
    ys[b][f] = z > 0.f ? z : 0.f;
  }
  __syncthreads();
  if (t < 64) {
    float z = fc2_b[0];
    for (int f = 0; f < 20; ++f) z += ys[t][f] * fc2_w[f];
    out_final[t] = 1.f / (1.f + expf(-z));
  }
}

// ----------------------------------------------------------------------------
extern "C" void kernel_launch(void* const* d_in, const int* in_sizes, int n_in,
                              void* d_out, int out_size, void* d_ws, size_t ws_size,
                              hipStream_t stream)
{
  // inputs (setup_inputs order; lists flattened)
  const float* x       = (const float*)d_in[0];
  const float* aw[8];  for (int i = 0; i < 8; ++i) aw[i] = (const float*)d_in[1 + i];
  const float* ab[8];  for (int i = 0; i < 8; ++i) ab[i] = (const float*)d_in[9 + i];
  const float* protos  = (const float*)d_in[17];
  const float* task_w  = (const float*)d_in[18];
  const float* task_b  = (const float*)d_in[19];
  const float* fc1_w   = (const float*)d_in[20];
  const float* fc1_b   = (const float*)d_in[21];
  const float* bn_g    = (const float*)d_in[22];
  const float* bn_b    = (const float*)d_in[23];
  const float* fc2_w   = (const float*)d_in[24];
  const float* fc2_b   = (const float*)d_in[25];

  // workspace layout
  char* ws = (char*)d_ws;
  const size_t SZ_BUF = (size_t)NPOS * 1024 * 2;        // 25,690,112
  _Float16* bufB    = (_Float16*)(ws);                  // ping buffer (also dist later)
  _Float16* bufC    = (_Float16*)(ws + SZ_BUF);         // pong buffer (final h lives here)
  _Float16* wbuf    = (_Float16*)(ws + 2 * SZ_BUF);     // f16 weights, 8,355,840 B
  _Float16* protosT = (_Float16*)(ws + 2 * SZ_BUF + 8355840);
  float*    p2      = (float*)   (ws + 2 * SZ_BUF + 8355840 + 65536);
  float*    x2      = (float*)   (ws + 2 * SZ_BUF + 8355840 + 65536 + 1024);
  float*    sim     = (float*)   (ws + 2 * SZ_BUF + 8355840 + 65536 + 1024 + 50176);
  float*    dist    = (float*)bufB;                     // aliases bufB after layer 8

  const int cin [8] = {2048, 1024, 1024, 512, 512, 256, 256, 128};
  const int cout[8] = {1024, 1024,  512, 512, 256, 256, 128, 128};
  _Float16* wp[8];
  {
    size_t off = 0;
    for (int i = 0; i < 8; ++i) { wp[i] = wbuf + off; off += (size_t)cout[i] * cin[i]; }
  }

  dim3 blk(256);

  // convert weights + prototypes to f16
  for (int i = 0; i < 8; ++i) {
    int n = cout[i] * cin[i];
    ppnet_cvt_f16<<<(n + 255) / 256, blk, 0, stream>>>(aw[i], wp[i], n);
  }
  ppnet_build_protos<<<(256 * 128) / 256, blk, 0, stream>>>(protos, protosT);
  ppnet_build_p2<<<1, blk, 0, stream>>>(protos, p2);

  // add-on bottleneck: 8 GEMMs, ping-pong B<->C, layer 0 reads x directly
  ppnet_gemm_wmma<0, true><<<dim3(98, cout[0] / 64), blk, 0, stream>>>(
      (const void*)x, wp[0], ab[0], (void*)bufB, nullptr, nullptr, cin[0], cout[0]);
  for (int i = 1; i < 7; ++i) {
    const void* src = (i % 2 == 1) ? (const void*)bufB : (const void*)bufC;
    void*       dst = (i % 2 == 0) ? (void*)bufB : (void*)bufC;
    ppnet_gemm_wmma<0, false><<<dim3(98, cout[i] / 64), blk, 0, stream>>>(
        src, wp[i], ab[i], dst, nullptr, nullptr, cin[i], cout[i]);
  }
  // layer 8: sigmoid, h -> bufC
  ppnet_gemm_wmma<1, false><<<dim3(98, cout[7] / 64), blk, 0, stream>>>(
      (const void*)bufB, wp[7], ab[7], (void*)bufC, nullptr, nullptr, cin[7], cout[7]);

  // ||h||^2 per position, then fused l2-distance GEMM (200 protos padded to 256 cols)
  ppnet_x2<<<(NPOS + 255) / 256, blk, 0, stream>>>(bufC, x2);
  ppnet_gemm_wmma<2, false><<<dim3(98, 4), blk, 0, stream>>>(
      (const void*)bufC, protosT, nullptr, (void*)dist, x2, p2, 128, 256);

  // outputs: final[64] | task_logits[1280] | min_d[12800]
  float* out       = (float*)d_out;
  float* out_final = out;
  float* out_task  = out + 64;
  float* out_mind  = out + 64 + 1280;

  ppnet_minpool<<<64, blk, 0, stream>>>(dist, sim, out_mind);
  ppnet_heads<<<1, blk, 0, stream>>>(sim, task_w, task_b, fc1_w, fc1_b,
                                     bn_g, bn_b, fc2_w, fc2_b, out_final, out_task);
}
